// STDPConv2d_18743237279788
// MI455X (gfx1250) — matrix operand order repaired and checked by reading, not compiled, not run on hardware
//
#include <hip/hip_runtime.h>
#include <hip/hip_bf16.h>

typedef __attribute__((ext_vector_type(16))) _Float16 v16h;
typedef __attribute__((ext_vector_type(8)))  _Float16 v8h;
typedef __attribute__((ext_vector_type(8)))  float    v8f;
typedef __attribute__((ext_vector_type(4)))  float    v4f;

union V16H { v16h v; v8h h[2]; };
union V8F  { v8f  v; v4f q[2]; };

#define IN_C   4
#define OUT_C  32
#define KS     5
#define PADP   2
#define BB     128
#define HH     84
#define WW     84
#define T_MAXF 15.0f
#define BASE_THRF 15.0f
#define KRED   100          // IN_C*KS*KS
#define KPAD   128          // 4 x K=32 WMMA chunks
#define AST    136          // LDS row stride (halves): 128 + 8 pad (272B, 16B aligned)
#define TW     16           // tile width  (= WMMA M)
#define THT    8            // tile height
#define TPIX   (TW*THT)     // 128 pixels / workgroup
#define NIMG   16           // images processed per workgroup (batched flush)

// ---------------------------------------------------------------------------
// Kernel 1: per (x,y) tile, loop over 16 images:
//   conv as implicit GEMM (v_wmma_f32_16x16x32_f16, b128 fragment loads)
//   -> WTA argmax -> dense one-hot latency output (b128 stores)
//   -> S_pre/count as a SECOND WMMA GEMM: OneHot[ch][px] x ActiveT[px][k],
//      accumulated in VGPRs across all 16 images, flushed once per WG.
// ---------------------------------------------------------------------------
__global__ __launch_bounds__(128) void stdp_conv_kernel(
    const float* __restrict__ lat, const float* __restrict__ wgt,
    const float* __restrict__ thr, float* __restrict__ out_lat,
    float* __restrict__ ws_cnt, float* __restrict__ ws_spre)
{
    __shared__ _Float16 s_in[IN_C * 12 * 20];                 // pot tile + halo
    __shared__ __align__(16) _Float16 s_amat[TPIX * AST];     // im2col [px][k]
    __shared__ __align__(16) _Float16 s_w[OUT_C * AST];       // weights [n][k]
    __shared__ __align__(16) unsigned char s_un[TPIX * AST * 2]; // union buffer
    __shared__ __align__(16) _Float16 s_oh[OUT_C * AST];      // one-hot [ch][px]
    __shared__ float s_thr[OUT_C];
    __shared__ float s_maxp[TPIX];
    __shared__ int   s_win[TPIX];

    float*    s_pot = (float*)s_un;        // [ch][px], live conv->argmax
    _Float16* s_at  = (_Float16*)s_un;     // act_T [k][px], live after argmax

    const int tid = threadIdx.x;
    const int x0 = blockIdx.x * TW;
    const int y0 = blockIdx.y * THT;
    const int b0 = blockIdx.z * NIMG;

    const int lane = tid & 31;
    const int wv   = tid >> 5;
    const int h    = lane >> 4;
    const int lm   = lane & 15;

    // persistent S_pre GEMM accumulators: D[ch][kcol], M=ch (2 tiles), N=kcol
    v8f sacc00 = {}, sacc01 = {}, sacc10 = {}, sacc11 = {};

    // ---- loop-invariant staging: weights (N-major, K zero-padded), thresholds
    for (int i = tid; i < OUT_C * KPAD; i += 128) {
        int n = i >> 7, k = i & 127;
        s_w[n * AST + k] = (k < KRED) ? (_Float16)wgt[n * KRED + k] : (_Float16)0.f;
    }
    if (tid < OUT_C) s_thr[tid] = thr[tid];

#pragma unroll 1
    for (int it = 0; it < NIMG; ++it) {
        const int b = b0 + it;

        // ---- phase A: stage input tile (pot) + zero one-hot ----
        for (int i = tid; i < IN_C * 12 * 20; i += 128) {
            int c  = i / 240, r = i % 240;
            int ry = r / 20,  rx = r % 20;
            int gy = y0 + ry - PADP, gx = x0 + rx - PADP;
            float p = 0.f;
            if (gy >= 0 && gy < HH && gx >= 0 && gx < WW) {
                float l = lat[((b * IN_C + c) * HH + gy) * WW + gx];
                p = (l > 0.f) ? (T_MAXF - l) : 0.f;
            }
            s_in[i] = (_Float16)p;
        }
        {
            v8h z = {};
            for (int i = tid; i < (OUT_C * AST) / 8; i += 128)
                *(v8h*)&s_oh[i * 8] = z;
        }
        __syncthreads();

        // ---- phase B: im2col row for this thread's pixel (const offsets) ----
        {
            const int rt = tid >> 4, xx = tid & 15;
            _Float16* dst = &s_amat[tid * AST];
            const _Float16* src = &s_in[rt * 20 + xx];
            v8h z = {};
#pragma unroll
            for (int q = 0; q < 5; ++q) *(v8h*)&dst[96 + 8 * q] = z;  // zero K pad
#pragma unroll
            for (int c = 0; c < IN_C; ++c)
#pragma unroll
                for (int ky = 0; ky < KS; ++ky)
#pragma unroll
                    for (int kx = 0; kx < KS; ++kx)
                        dst[c * 25 + ky * 5 + kx] = src[c * 240 + ky * 20 + kx];
        }
        __syncthreads();

        // ---- phase C: conv WMMA (A = im2col rows, B = weights) ----
        {
            V16H bf[2][4];
#pragma unroll
            for (int nt = 0; nt < 2; ++nt)
#pragma unroll
                for (int kc = 0; kc < 4; ++kc) {
                    const _Float16* bp = &s_w[(nt * 16 + lm) * AST + kc * 32 + 8 * h];
                    bf[nt][kc].h[0] = *(const v8h*)bp;
                    bf[nt][kc].h[1] = *(const v8h*)(bp + 16);
                }
#pragma unroll
            for (int rr = 0; rr < 2; ++rr) {
                const int row = wv * 2 + rr;
                const _Float16* ap = &s_amat[(row * TW + lm) * AST];
                v8f acc0 = {}; v8f acc1 = {};
#pragma unroll
                for (int kc = 0; kc < 4; ++kc) {
                    V16H a;
                    a.h[0] = *(const v8h*)(ap + kc * 32 + 8 * h);
                    a.h[1] = *(const v8h*)(ap + kc * 32 + 16 + 8 * h);
                    acc0 = __builtin_amdgcn_wmma_f32_16x16x32_f16(false, a.v, false, bf[0][kc].v,
                                                                  (short)0, acc0, false, false);
                    acc1 = __builtin_amdgcn_wmma_f32_16x16x32_f16(false, a.v, false, bf[1][kc].v,
                                                                  (short)0, acc1, false, false);
                }
                const int pbase = row * TW + 8 * h;
                V8F u0; u0.v = acc0;
                V8F u1; u1.v = acc1;
                *(v4f*)&s_pot[lm * TPIX + pbase]            = u0.q[0];
                *(v4f*)&s_pot[lm * TPIX + pbase + 4]        = u0.q[1];
                *(v4f*)&s_pot[(lm + 16) * TPIX + pbase]     = u1.q[0];
                *(v4f*)&s_pot[(lm + 16) * TPIX + pbase + 4] = u1.q[1];
            }
        }
        __syncthreads();

        // ---- phase D: per-pixel argmax + one-hot scatter ----
        {
            const int p  = tid;
            const int rt = p >> 4, xx = p & 15;
            const int gy = y0 + rt, gx = x0 + xx;
            const bool inb = (gy < HH && gx < WW);

            float mp = s_pot[p]; int wn = 0;
#pragma unroll
            for (int n = 1; n < OUT_C; ++n) {
                float v = s_pot[n * TPIX + p];
                if (v > mp) { mp = v; wn = n; }   // first-max tie-break
            }
            s_maxp[p] = mp; s_win[p] = wn;
            const bool fire = inb && (mp >= s_thr[wn]);
            s_oh[wn * AST + p] = fire ? (_Float16)1.f : (_Float16)0.f;
        }
        __syncthreads();   // s_pot dead from here -> reuse as act_T

        // ---- phase E: stage act_T[k][px] (+ count column at k=100) and
        //               write dense one-hot latency output (float4) ----
        {
            const int kcol = tid;                        // 128 threads = 128 k-cols
            const _Float16* colp = &s_amat[kcol];        // stride AST
            _Float16* dstc = &s_at[kcol * AST];
            const _Float16 onev = (_Float16)1.f;
#pragma unroll 2
            for (int g = 0; g < 16; ++g) {
                v8h v;
#pragma unroll
                for (int u = 0; u < 8; ++u) {
                    float a = (float)colp[(g * 8 + u) * AST];
                    v[u] = (a > 0.f || kcol == KRED) ? onev : (_Float16)0.f;
                }
                *(v8h*)&dstc[g * 8] = v;
            }
        }
        for (int j = 0; j < (OUT_C * TPIX) / (4 * 128); ++j) {   // 8 iters
            int q  = j * 128 + tid;
            int x4 = (q & 3) * 4;
            int rt = (q >> 2) & 7;
            int n  = q >> 5;
            int gy = y0 + rt, gx = x0 + x4;
            if (gy < HH && gx < WW) {                  // 84 % 4 == 0: group all-in/out
                v4f o;
#pragma unroll
                for (int u = 0; u < 4; ++u) {
                    int p = rt * TW + x4 + u;
                    float mp = s_maxp[p]; int wn = s_win[p];
                    bool fire = (mp >= s_thr[wn]);
                    o[u] = (fire && n == wn) ? (T_MAXF * BASE_THRF) / (mp + 1e-5f) : 0.f;
                }
                *(v4f*)&out_lat[((b * OUT_C + n) * HH + gy) * WW + gx] = o;
            }
        }
        __syncthreads();

        // ---- phase F: S_pre GEMM accumulate: OneHot[ch][px] x act_T[px][kcol]
        //      wave wv owns kcol tiles {2wv, 2wv+1}; M = ch (2 tiles); K = px.
#pragma unroll
        for (int kc = 0; kc < 4; ++kc) {
            V16H a0, a1, bb0, bb1;
            const _Float16* ap0 = &s_oh[lm * AST + kc * 32 + 8 * h];
            const _Float16* ap1 = &s_oh[(16 + lm) * AST + kc * 32 + 8 * h];
            a0.h[0] = *(const v8h*)ap0;
            a0.h[1] = *(const v8h*)(ap0 + 16);
            a1.h[0] = *(const v8h*)ap1;
            a1.h[1] = *(const v8h*)(ap1 + 16);
            const _Float16* bp0 = &s_at[((2 * wv + 0) * 16 + lm) * AST + kc * 32 + 8 * h];
            const _Float16* bp1 = &s_at[((2 * wv + 1) * 16 + lm) * AST + kc * 32 + 8 * h];
            bb0.h[0] = *(const v8h*)bp0;  bb0.h[1] = *(const v8h*)(bp0 + 16);
            bb1.h[0] = *(const v8h*)bp1;  bb1.h[1] = *(const v8h*)(bp1 + 16);
            sacc00 = __builtin_amdgcn_wmma_f32_16x16x32_f16(false, a0.v, false, bb0.v,
                                                            (short)0, sacc00, false, false);
            sacc01 = __builtin_amdgcn_wmma_f32_16x16x32_f16(false, a0.v, false, bb1.v,
                                                            (short)0, sacc01, false, false);
            sacc10 = __builtin_amdgcn_wmma_f32_16x16x32_f16(false, a1.v, false, bb0.v,
                                                            (short)0, sacc10, false, false);
            sacc11 = __builtin_amdgcn_wmma_f32_16x16x32_f16(false, a1.v, false, bb1.v,
                                                            (short)0, sacc11, false, false);
        }
        __syncthreads();   // protect s_oh/s_at/s_amat rewrite next iteration
    }

    // ---- final flush: one pass per WG. kcol<100 -> S_pre; kcol==100 -> count.
    {
        auto flush = [&](v8f acc, int m, int nn) {
            const int kcol = (2 * wv + nn) * 16 + lm;
#pragma unroll
            for (int r = 0; r < 8; ++r) {
                float v = acc[r];
                if (v != 0.f) {
                    int ch = m * 16 + r + 8 * h;
                    if (kcol < KRED)       atomicAdd(&ws_spre[ch * KRED + kcol], v);
                    else if (kcol == KRED) atomicAdd(&ws_cnt[ch], v);
                }
            }
        };
        flush(sacc00, 0, 0); flush(sacc01, 0, 1);
        flush(sacc10, 1, 0); flush(sacc11, 1, 1);
    }
}

// ---------------------------------------------------------------------------
// Kernel 2: homeostasis (new_ema, new_thr).
// ---------------------------------------------------------------------------
__global__ void stdp_homeo_kernel(const float* __restrict__ thr,
                                  const float* __restrict__ ema,
                                  const float* __restrict__ ws_cnt,
                                  float* __restrict__ out_thr,
                                  float* __restrict__ out_ema)
{
    int n = threadIdx.x;
    if (n < OUT_C) {
        float cnt  = ws_cnt[n];
        float rate = cnt / (903168.f + 1e-8f);       // B*H*W
        float ne   = 0.99f * ema[n] + 0.01f * rate;
        float err  = ne - 0.05f;
        float nt   = thr[n] + 0.01f * err * thr[n];
        nt = fminf(fmaxf(nt, BASE_THRF * 0.1f), BASE_THRF * 5.0f);
        out_thr[n] = nt;
        out_ema[n] = ne;
    }
}

// ---------------------------------------------------------------------------
// Kernel 3: STDP weight update + per-channel L2 renorm (8 threads/channel).
// ---------------------------------------------------------------------------
__global__ __launch_bounds__(256) void stdp_wupd_kernel(
    const float* __restrict__ wgt, const float* __restrict__ ws_cnt,
    const float* __restrict__ ws_spre, float* __restrict__ out_w)
{
    __shared__ float red[256];
    __shared__ float red2[256];
    __shared__ float s_chmin[OUT_C], s_chmax[OUT_C], s_norm[OUT_C];
    __shared__ float s_wmin, s_wmax;
    __shared__ int   s_anyf;
    __shared__ float s_wu[OUT_C * KRED];

    const int tid = threadIdx.x;
    const int ch  = tid >> 3, sub = tid & 7;

    float mn = 1e30f, mx = -1e30f;
    for (int k = sub; k < KRED; k += 8) {
        float w = wgt[ch * KRED + k];
        mn = fminf(mn, w); mx = fmaxf(mx, w);
    }
    red[tid] = mn; red2[tid] = mx;
    __syncthreads();
    if (sub == 0) {
        for (int s = 1; s < 8; ++s) {
            mn = fminf(mn, red[tid + s]); mx = fmaxf(mx, red2[tid + s]);
        }
        s_chmin[ch] = mn; s_chmax[ch] = mx;
    }
    __syncthreads();
    if (tid == 0) {
        float wmin = 1e30f, wmax = -1e30f; int anyf = 0;
        for (int n = 0; n < OUT_C; ++n) {
            if (ws_cnt[n] > 0.f) {
                anyf = 1;
                wmin = fminf(wmin, s_chmin[n]);
                wmax = fmaxf(wmax, s_chmax[n]);
            }
        }
        s_wmin = wmin; s_wmax = wmax; s_anyf = anyf;
    }
    __syncthreads();

    const float wmin = s_wmin, wmax = s_wmax;
    const float inv  = 1.f / (wmax - wmin + 1e-6f);
    const float cnt  = ws_cnt[ch];
    const int fired  = cnt > 0.f;
    const float cden = 1.f / fmaxf(cnt, 1.f);
    float ssq = 0.f;
    for (int k = sub; k < KRED; k += 8) {
        int i = ch * KRED + k;
        float w0 = wgt[i];
        float wn = (w0 - wmin) * inv;
        float sp = ws_spre[i];
        float d  = 0.004f * (1.f - wn) * sp - 0.0012f * wn * (cnt - sp);
        d = fired ? d * cden : 0.f;
        float wu = fmaxf(w0 + d, 0.01f);
        s_wu[i] = wu;
        ssq += wu * wu;
    }
    red[tid] = ssq;
    __syncthreads();
    if (sub == 0) {
        float s = 0.f;
        for (int q = 0; q < 8; ++q) s += red[ch * 8 + q];
        s_norm[ch] = sqrtf(s);
    }
    __syncthreads();
    const float scale = 10.f / (s_norm[ch] + 1e-6f);   // FAN_IN_SQRT = 10
    const int anyf = s_anyf;
    for (int k = sub; k < KRED; k += 8) {
        int i = ch * KRED + k;
        out_w[i] = anyf ? s_wu[i] * scale : wgt[i];
    }
}

// ---------------------------------------------------------------------------
// Workspace zero-fill (graph-capture safe).
// ---------------------------------------------------------------------------
__global__ void zero_ws_kernel(float* __restrict__ p, int n)
{
    int i = blockIdx.x * blockDim.x + threadIdx.x;
    if (i < n) p[i] = 0.f;
}

extern "C" void kernel_launch(void* const* d_in, const int* in_sizes, int n_in,
                              void* d_out, int out_size, void* d_ws, size_t ws_size,
                              hipStream_t stream) {
    (void)in_sizes; (void)n_in; (void)out_size; (void)ws_size;
    const float* lat = (const float*)d_in[0];   // (128,4,84,84)
    const float* wgt = (const float*)d_in[1];   // (32,4,5,5)
    const float* thr = (const float*)d_in[2];   // (32,)
    const float* ema = (const float*)d_in[3];   // (32,)

    float* out      = (float*)d_out;
    float* out_lat  = out;                                  // (128,32,84,84)
    float* out_w    = out + (size_t)BB * OUT_C * HH * WW;   // (32,4,5,5)
    float* out_thr  = out_w + OUT_C * KRED;                 // (32,)
    float* out_ema  = out_thr + OUT_C;                      // (32,)

    float* ws      = (float*)d_ws;
    float* ws_cnt  = ws;                 // 32 floats
    float* ws_spre = ws + OUT_C;         // 3200 floats

    const int zn = OUT_C + OUT_C * KRED;
    zero_ws_kernel<<<(zn + 255) / 256, 256, 0, stream>>>(ws, zn);

    dim3 grid((WW + TW - 1) / TW, (HH + THT - 1) / THT, BB / NIMG);   // 6 x 11 x 8
    stdp_conv_kernel<<<grid, 128, 0, stream>>>(lat, wgt, thr, out_lat, ws_cnt, ws_spre);

    stdp_homeo_kernel<<<1, 32, 0, stream>>>(thr, ema, ws_cnt, out_thr, out_ema);
    stdp_wupd_kernel<<<1, 256, 0, stream>>>(wgt, ws_cnt, ws_spre, out_w);
}
